// FlashAttn_38517266710795
// MI455X (gfx1250) — matrix operand order repaired and checked by reading, not compiled, not run on hardware
//
#include <hip/hip_runtime.h>
#include <math.h>

typedef __attribute__((ext_vector_type(16))) __bf16 v16bf;
typedef __attribute__((ext_vector_type(8)))  __bf16 v8bf;
typedef __attribute__((ext_vector_type(8)))  float  v8f;

#define S_LEN 2048
#define D_DIM 128
#define BM    128                 // query rows per block (8 waves x 16 rows)
#define BN    64                  // keys per tile iteration
#define NWAVE 8
#define NTILES (S_LEN / BN)       // 32
#define LOG2E  1.44269504088896340736f

static __device__ __forceinline__ v8f wmma_bf16(v16bf a, v16bf b, v8f c) {
    return __builtin_amdgcn_wmma_f32_16x16x32_bf16(false, a, false, b,
                                                   (short)0, c, false, false);
}

__global__ __launch_bounds__(256, 1)
void fattn_wmma_bf16(const float* __restrict__ q,
                     const float* __restrict__ k,
                     const float* __restrict__ v,
                     float* __restrict__ o)
{
    // Double-buffered K/V tiles + per-wave P scratch (C-layout -> A-layout).
    __shared__ __bf16 kt[2][BN][D_DIM];        // 2 x 16 KB, row-major [key][d]
    __shared__ __bf16 vt[2][D_DIM][BN];        // 2 x 16 KB, transposed [d][key]
    __shared__ __bf16 pbuf[NWAVE][16][BN];     // 16 KB

    const int tid  = threadIdx.x;
    const int wave = tid >> 5;
    const int lane = tid & 31;
    const int l16  = lane & 15;
    const int hi   = lane >> 4;      // lane half (A/B layout K-split)

    const int bid   = blockIdx.x;
    const int qtile = bid & 15;      // S / BM = 16 tiles
    const int bh    = bid >> 4;      // B*H = 32

    const float* qp = q + (size_t)bh * S_LEN * D_DIM;
    const float* kp = k + (size_t)bh * S_LEN * D_DIM;
    const float* vp = v + (size_t)bh * S_LEN * D_DIM;
    float*       op = o + (size_t)bh * S_LEN * D_DIM;

    const int qbase = qtile * BM + wave * 16;
    const int qrow  = qbase + l16;   // A-matrix row held by this lane

    // ---- Q -> A-fragments, pre-scaled by log2(e) (scores in log2 domain) ----
    // A 16x32 bf16: lanes 0-15 hold K {0..15}, lanes 16-31 {16..31}
    v16bf qa[4];
#pragma unroll
    for (int c = 0; c < 4; ++c) {
        const float* b0 = qp + (size_t)qrow * D_DIM + c * 32 + hi * 16;
        const float* b1 = b0 + 8;
#pragma unroll
        for (int e = 0; e < 8; ++e) {
            qa[c][e]     = (__bf16)(b0[e] * LOG2E);
            qa[c][e + 8] = (__bf16)(b1[e] * LOG2E);
        }
    }

    v8f acc_o[8];                    // 8 D-tiles, C-layout (row = j + 8*hi)
#pragma unroll
    for (int i = 0; i < 8; ++i)
        acc_o[i] = (v8f){0.f,0.f,0.f,0.f,0.f,0.f,0.f,0.f};

    float mrun[8];
#pragma unroll
    for (int j = 0; j < 8; ++j) mrun[j] = -__builtin_inff();

    // ---- tile staging helpers (register prefetch -> convert -> LDS) ----
    float4 kreg[8], vreg[8];
    auto load_tile = [&](int kb) {
#pragma unroll
        for (int i = 0; i < 8; ++i) {
            const int f = (tid + i * 256) * 4;     // flat elem idx in 64x128
            kreg[i] = *(const float4*)(kp + (size_t)kb * D_DIM + f);
            vreg[i] = *(const float4*)(vp + (size_t)kb * D_DIM + f);
        }
    };
    auto store_tile = [&](int buf) {
#pragma unroll
        for (int i = 0; i < 8; ++i) {
            const int f = (tid + i * 256) * 4;
            const int row = f >> 7, col = f & 127;
            __bf16* dk = &kt[buf][row][col];
            dk[0] = (__bf16)kreg[i].x; dk[1] = (__bf16)kreg[i].y;
            dk[2] = (__bf16)kreg[i].z; dk[3] = (__bf16)kreg[i].w;
            vt[buf][col + 0][row] = (__bf16)vreg[i].x;
            vt[buf][col + 1][row] = (__bf16)vreg[i].y;
            vt[buf][col + 2][row] = (__bf16)vreg[i].z;
            vt[buf][col + 3][row] = (__bf16)vreg[i].w;
        }
    };

    // ---- prologue: stage tile 0 into buffer 0 ----
    load_tile(0);
    store_tile(0);

    for (int it = 0; it < NTILES; ++it) {
        const int cur = it & 1;
        // One barrier/iter: covers RAW on buf[cur] (written end of it-1) and
        // WAR on buf[cur^1] (read during it-1, rewritten at end of this iter).
        __syncthreads();

        if (it + 1 < NTILES) {
            load_tile((it + 1) * BN);              // async until consumed
            if (it + 2 < NTILES) {                 // L2 prefetch 2 tiles ahead
                const size_t pfo = (size_t)(it + 2) * BN * D_DIM + tid * 32;
                __builtin_prefetch(kp + pfo, 0, 1);
                __builtin_prefetch(vp + pfo, 0, 1);
            }
        }

        // ---- S' = (Q*log2e) K^T : K-chunk outer, 4 INDEPENDENT WMMAs ----
        v8f s[4];
#pragma unroll
        for (int t = 0; t < 4; ++t)
            s[t] = (v8f){0.f,0.f,0.f,0.f,0.f,0.f,0.f,0.f};
#pragma unroll
        for (int ch = 0; ch < 4; ++ch) {
            // B 32x16 bf16: lane holds col l16, K = ch*32 + hi*16 + (0..15)
            const v16bf b0 = *(const v16bf*)&kt[cur][ 0 + l16][ch * 32 + hi * 16];
            const v16bf b1 = *(const v16bf*)&kt[cur][16 + l16][ch * 32 + hi * 16];
            const v16bf b2 = *(const v16bf*)&kt[cur][32 + l16][ch * 32 + hi * 16];
            const v16bf b3 = *(const v16bf*)&kt[cur][48 + l16][ch * 32 + hi * 16];
            s[0] = wmma_bf16(qa[ch], b0, s[0]);
            s[1] = wmma_bf16(qa[ch], b1, s[1]);
            s[2] = wmma_bf16(qa[ch], b2, s[2]);
            s[3] = wmma_bf16(qa[ch], b3, s[3]);
        }

        // ---- online row max (log2 domain) ----
        float mnew[8], scale[8];
#pragma unroll
        for (int j = 0; j < 8; ++j) {
            float m = fmaxf(fmaxf(s[0][j], s[1][j]), fmaxf(s[2][j], s[3][j]));
#pragma unroll
            for (int off = 1; off < 16; off <<= 1)
                m = fmaxf(m, __shfl_xor(m, off, 16));
            mnew[j]  = fmaxf(mrun[j], m);
            scale[j] = exp2f(mrun[j] - mnew[j]);   // first iter: exp2(-inf)=0
            mrun[j]  = mnew[j];
        }

        // ---- P = exp2(S' - m'), spill to wave-private LDS [16][64] ----
#pragma unroll
        for (int t = 0; t < 4; ++t)
#pragma unroll
            for (int j = 0; j < 8; ++j) {
                const float pv = exp2f(s[t][j] - mnew[j]);
                pbuf[wave][8 * hi + j][t * 16 + l16] = (__bf16)pv;
            }

        asm volatile("s_wait_dscnt 0" ::: "memory");   // wave-local RAW on pbuf

        // ---- rescale running output ----
#pragma unroll
        for (int dt = 0; dt < 8; ++dt)
#pragma unroll
            for (int j = 0; j < 8; ++j)
                acc_o[dt][j] *= scale[j];

        // ---- O += P V : 2 K-chunks x 8 independent accumulators ----
#pragma unroll
        for (int ch = 0; ch < 2; ++ch) {
            const v8bf a0 = *(const v8bf*)&pbuf[wave][l16][ch * 32 + hi * 16];
            const v8bf a1 = *(const v8bf*)&pbuf[wave][l16][ch * 32 + hi * 16 + 8];
            const v16bf a = __builtin_shufflevector(a0, a1,
                0,1,2,3,4,5,6,7,8,9,10,11,12,13,14,15);
#pragma unroll
            for (int dt = 0; dt < 8; ++dt) {
                const v16bf b = *(const v16bf*)&vt[cur][dt * 16 + l16][ch * 32 + hi * 16];
                acc_o[dt] = wmma_bf16(a, b, acc_o[dt]);
            }
        }

        // ---- stage next tile into the other buffer (loads had all iter) ----
        if (it + 1 < NTILES) store_tile(cur ^ 1);
    }

    // ---- write unnormalized O (fp32), C-layout row = j + 8*hi ----
#pragma unroll
    for (int dt = 0; dt < 8; ++dt)
#pragma unroll
        for (int j = 0; j < 8; ++j) {
            const int row = qbase + 8 * hi + j;
            op[(size_t)row * D_DIM + dt * 16 + l16] = acc_o[dt][j];
        }
}

extern "C" void kernel_launch(void* const* d_in, const int* in_sizes, int n_in,
                              void* d_out, int out_size, void* d_ws, size_t ws_size,
                              hipStream_t stream) {
    (void)in_sizes; (void)n_in; (void)out_size; (void)d_ws; (void)ws_size;
    const float* q = (const float*)d_in[0];
    const float* k = (const float*)d_in[1];
    const float* v = (const float*)d_in[2];
    float* o = (float*)d_out;
    // grid = (B*H) * (S/BM) = 32 * 16 = 512 blocks, 256 threads (8 wave32s)
    fattn_wmma_bf16<<<dim3(512), dim3(256), 0, stream>>>(q, k, v, o);
}
